// HiSumNet_86397562126549
// MI455X (gfx1250) — compile-verified, matrix operand order checked by reading, not run on hardware
//
#include <hip/hip_runtime.h>
#include <hip/hip_bf16.h>
#include <math.h>

typedef __attribute__((ext_vector_type(16))) __bf16 v16bf;
typedef __attribute__((ext_vector_type(8)))  __bf16 v8bf;
typedef __attribute__((ext_vector_type(8)))  float  v8f;

#define DMODEL 1024
#define NHEAD  16
#define HD     64
#define WS_STRIDE 132   // 128 cols + 4 pad floats: K-offset of 8 rows -> +32 banks

__device__ __forceinline__ float gelu_f(float x) {
    float x3 = x * x * x;
    return 0.5f * x * (1.0f + tanhf(0.7978845608028654f * (x + 0.044715f * x3)));
}

// ---------------------------------------------------------------- seg ids ---
__global__ void seg_kernel(const int* __restrict__ bd, int* __restrict__ seg,
                           int T, int nb) {
    int b = blockIdx.x;
    const int* bb = bd + b * nb;
    for (int t = threadIdx.x; t < T; t += blockDim.x) {
        int c = 0;
        for (int j = 0; j < nb; ++j) c += (bb[j] <= t) ? 1 : 0;   // searchsorted right
        seg[b * T + t] = c;
    }
}

// ------------------------------------------------------- segment mean agg ---
__global__ void aggregate_kernel(const float* __restrict__ X, const int* __restrict__ bd,
                                 float* __restrict__ out, int T, int S, int nb) {
    int b = blockIdx.y, s = blockIdx.x;
    int t0 = (s == 0) ? 0 : bd[b * nb + s - 1];
    int t1 = (s == S - 1) ? T : bd[b * nb + s];
    if (t1 <= t0) t1 = t0 + 1;
    float inv = 1.0f / (float)(t1 - t0);
    const float* Xb = X + (size_t)b * T * DMODEL;
    float* ob = out + ((size_t)b * S + s) * DMODEL;
    for (int d = threadIdx.x; d < DMODEL; d += blockDim.x) {
        float acc = 0.f;
        for (int t = t0; t < t1; ++t) acc += Xb[(size_t)t * DMODEL + d];
        ob[d] = acc * inv;
    }
}

// --------------------------------------------------------------- layernorm --
__global__ void layernorm_kernel(const float* __restrict__ X, const float* __restrict__ g,
                                 const float* __restrict__ bta, float* __restrict__ Y) {
    __shared__ float s1[256], s2[256];
    int row = blockIdx.x, tid = threadIdx.x;
    const float* xr = X + (size_t)row * DMODEL;
    float sum = 0.f, sq = 0.f;
    for (int d = tid; d < DMODEL; d += 256) { float v = xr[d]; sum += v; sq += v * v; }
    s1[tid] = sum; s2[tid] = sq;
    __syncthreads();
    for (int off = 128; off > 0; off >>= 1) {
        if (tid < off) { s1[tid] += s1[tid + off]; s2[tid] += s2[tid + off]; }
        __syncthreads();
    }
    float mu  = s1[0] * (1.0f / DMODEL);
    float var = s2[0] * (1.0f / DMODEL) - mu * mu;
    float rstd = rsqrtf(var + 1e-5f);
    float* yr = Y + (size_t)row * DMODEL;
    for (int d = tid; d < DMODEL; d += 256)
        yr[d] = (xr[d] - mu) * rstd * g[d] + bta[d];
}

// ------------------------------------------------- bf16 WMMA GEMM + epilog --
// C[M,N] = act(A[M,K] @ W[K,N] + bias) (+ residual).
// Block: 8 waves -> 64(M) x 128(N) tile; wave = 16 x 64 (4 accums), K step 32.
// W tile (32x128 fp32) is async-DMA'd to LDS (double buffered, prefetch 1 ahead).
__global__ void gemm_wmma_kernel(const float* __restrict__ A, const float* __restrict__ W,
                                 const float* __restrict__ bias, const float* __restrict__ res,
                                 float* __restrict__ C, int M, int K, int N, int act) {
    __shared__ float Ws[2][32 * WS_STRIDE];

    int tid  = threadIdx.x;
    int lane = tid & 31;
    int wave = tid >> 5;
    int wm = wave >> 1, wn = wave & 1;
    int l16 = lane & 15, h = lane >> 4;
    int row0 = blockIdx.y * 64 + wm * 16;
    int col0 = blockIdx.x * 128;

    unsigned wsOff[2];
    wsOff[0] = (unsigned)(size_t)(&Ws[0][0]);
    wsOff[1] = (unsigned)(size_t)(&Ws[1][0]);

    // one wave-instruction copies one 512B W row slice (32 lanes x 16B)
    auto issueW = [&](int kt, int buf) {
        int k0 = kt * 32;
        #pragma unroll
        for (int i = 0; i < 4; ++i) {
            int krow = i * 8 + wave;                       // 0..31
            const float* gp = W + (size_t)(k0 + krow) * N + col0 + lane * 4;
            unsigned la = wsOff[buf] + (unsigned)(krow * WS_STRIDE + lane * 4) * 4u;
            asm volatile("global_load_async_to_lds_b128 %0, %1, off"
                         :: "v"(la), "v"(gp) : "memory");
        }
    };

    v8f z = {};
    v8f acc[4] = {z, z, z, z};
    const float* Ar = A + (size_t)(row0 + l16) * K;
    int nk = K >> 5;

    issueW(0, 0);
    for (int kt = 0; kt < nk; ++kt) {
        int buf = kt & 1;
        if (kt + 1 < nk) {
            issueW(kt + 1, buf ^ 1);
            asm volatile("s_wait_asynccnt 0x4" ::: "memory");
        } else {
            asm volatile("s_wait_asynccnt 0x0" ::: "memory");
        }
        __syncthreads();   // tile kt visible to all waves

        int k0 = kt * 32;
        // A fragment: per-lane contiguous 8-float runs -> b128 loads
        v16bf a;
        {
            float4 q0 = *(const float4*)(Ar + k0 + 8 * h);
            float4 q1 = *(const float4*)(Ar + k0 + 8 * h + 4);
            float4 q2 = *(const float4*)(Ar + k0 + 16 + 8 * h);
            float4 q3 = *(const float4*)(Ar + k0 + 16 + 8 * h + 4);
            a[0] = (__bf16)q0.x; a[1] = (__bf16)q0.y; a[2] = (__bf16)q0.z; a[3] = (__bf16)q0.w;
            a[4] = (__bf16)q1.x; a[5] = (__bf16)q1.y; a[6] = (__bf16)q1.z; a[7] = (__bf16)q1.w;
            a[8]  = (__bf16)q2.x; a[9]  = (__bf16)q2.y; a[10] = (__bf16)q2.z; a[11] = (__bf16)q2.w;
            a[12] = (__bf16)q3.x; a[13] = (__bf16)q3.y; a[14] = (__bf16)q3.z; a[15] = (__bf16)q3.w;
        }
        const float* wsb = &Ws[buf][0];
        #pragma unroll
        for (int t = 0; t < 4; ++t) {
            const float* wc = wsb + wn * 64 + t * 16 + l16;   // column in LDS tile
            v16bf bf;
            #pragma unroll
            for (int e = 0; e < 8; ++e) bf[e]     = (__bf16)wc[(8 * h + e) * WS_STRIDE];
            #pragma unroll
            for (int e = 0; e < 8; ++e) bf[8 + e] = (__bf16)wc[(16 + 8 * h + e) * WS_STRIDE];
            acc[t] = __builtin_amdgcn_wmma_f32_16x16x32_bf16(
                false, a, false, bf, (short)0, acc[t], false, false);
        }
        __syncthreads();   // done reading tile kt before its buffer is rewritten
    }

    #pragma unroll
    for (int t = 0; t < 4; ++t) {
        int col = col0 + wn * 64 + t * 16 + l16;
        float bv = bias[col];
        #pragma unroll
        for (int r = 0; r < 8; ++r) {
            int row = row0 + r + 8 * h;           // C/D: M = r + 8*(lane/16)
            float v = acc[t][r] + bv;
            if (act) v = gelu_f(v);
            size_t idx = (size_t)row * N + col;
            if (res) v += res[idx];
            C[idx] = v;
        }
    }
}

// ------------------------------------------------------ fused attention -----
// One wave per (q-tile of 16, head, batch). qkv: [B,S,3*D], q|k|v slices of D.
template <int S>
__global__ void attention_kernel(const float* __restrict__ qkv, float* __restrict__ out) {
    constexpr int NT = S / 16;
    __shared__ __align__(16) __bf16 Pl[16 * S];
    int lane = threadIdx.x & 31;
    int l16 = lane & 15, h = lane >> 4;
    int qt = blockIdx.x, hh = blockIdx.y, b = blockIdx.z;
    const float* base = qkv + (size_t)b * S * (3 * DMODEL);

    v8f z = {};
    v8f sc[NT];
    #pragma unroll
    for (int i = 0; i < NT; ++i) sc[i] = z;

    // scores = Q @ K^T
    const float* qrow = base + (size_t)(qt * 16 + l16) * (3 * DMODEL) + hh * HD;
    #pragma unroll
    for (int k0 = 0; k0 < HD; k0 += 32) {
        v16bf a;
        {
            float4 q0 = *(const float4*)(qrow + k0 + 8 * h);
            float4 q1 = *(const float4*)(qrow + k0 + 8 * h + 4);
            float4 q2 = *(const float4*)(qrow + k0 + 16 + 8 * h);
            float4 q3 = *(const float4*)(qrow + k0 + 16 + 8 * h + 4);
            a[0] = (__bf16)q0.x; a[1] = (__bf16)q0.y; a[2] = (__bf16)q0.z; a[3] = (__bf16)q0.w;
            a[4] = (__bf16)q1.x; a[5] = (__bf16)q1.y; a[6] = (__bf16)q1.z; a[7] = (__bf16)q1.w;
            a[8]  = (__bf16)q2.x; a[9]  = (__bf16)q2.y; a[10] = (__bf16)q2.z; a[11] = (__bf16)q2.w;
            a[12] = (__bf16)q3.x; a[13] = (__bf16)q3.y; a[14] = (__bf16)q3.z; a[15] = (__bf16)q3.w;
        }
        #pragma unroll
        for (int kt = 0; kt < NT; ++kt) {
            const float* krow = base + (size_t)(kt * 16 + l16) * (3 * DMODEL) + DMODEL + hh * HD;
            v16bf bf;
            float4 k0v = *(const float4*)(krow + k0 + 8 * h);
            float4 k1v = *(const float4*)(krow + k0 + 8 * h + 4);
            float4 k2v = *(const float4*)(krow + k0 + 16 + 8 * h);
            float4 k3v = *(const float4*)(krow + k0 + 16 + 8 * h + 4);
            bf[0] = (__bf16)k0v.x; bf[1] = (__bf16)k0v.y; bf[2] = (__bf16)k0v.z; bf[3] = (__bf16)k0v.w;
            bf[4] = (__bf16)k1v.x; bf[5] = (__bf16)k1v.y; bf[6] = (__bf16)k1v.z; bf[7] = (__bf16)k1v.w;
            bf[8]  = (__bf16)k2v.x; bf[9]  = (__bf16)k2v.y; bf[10] = (__bf16)k2v.z; bf[11] = (__bf16)k2v.w;
            bf[12] = (__bf16)k3v.x; bf[13] = (__bf16)k3v.y; bf[14] = (__bf16)k3v.z; bf[15] = (__bf16)k3v.w;
            sc[kt] = __builtin_amdgcn_wmma_f32_16x16x32_bf16(
                false, a, false, bf, (short)0, sc[kt], false, false);
        }
    }

    // row softmax (mask all-true); row r+8h lives in elem r across a 16-lane group
    const float scale = 0.125f;  // 1/sqrt(64)
    #pragma unroll
    for (int r = 0; r < 8; ++r) {
        float m = -1e30f;
        #pragma unroll
        for (int kt = 0; kt < NT; ++kt) m = fmaxf(m, sc[kt][r] * scale);
        #pragma unroll
        for (int off = 8; off >= 1; off >>= 1) m = fmaxf(m, __shfl_xor(m, off, 32));
        float ssum = 0.f;
        #pragma unroll
        for (int kt = 0; kt < NT; ++kt) {
            float p = __expf(sc[kt][r] * scale - m);
            sc[kt][r] = p;
            ssum += p;
        }
        #pragma unroll
        for (int off = 8; off >= 1; off >>= 1) ssum += __shfl_xor(ssum, off, 32);
        float rinv = 1.0f / ssum;
        #pragma unroll
        for (int kt = 0; kt < NT; ++kt) sc[kt][r] *= rinv;
    }

    // bounce P through LDS: C-layout -> A-layout, as bf16
    #pragma unroll
    for (int kt = 0; kt < NT; ++kt)
        #pragma unroll
        for (int r = 0; r < 8; ++r)
            Pl[(r + 8 * h) * S + kt * 16 + l16] = (__bf16)sc[kt][r];
    asm volatile("s_wait_dscnt 0x0" ::: "memory");

    // O = P @ V
    v8f oc[4] = {z, z, z, z};
    for (int k0 = 0; k0 < S; k0 += 32) {
        v16bf a;
        {
            v8bf c0 = *(const v8bf*)(&Pl[l16 * S + k0 + 8 * h]);
            v8bf c1 = *(const v8bf*)(&Pl[l16 * S + k0 + 16 + 8 * h]);
            #pragma unroll
            for (int e = 0; e < 8; ++e) { a[e] = c0[e]; a[8 + e] = c1[e]; }
        }
        #pragma unroll
        for (int nt = 0; nt < 4; ++nt) {
            const float* v0 = base + (size_t)(k0 + 8 * h) * (3 * DMODEL) + 2 * DMODEL + hh * HD + nt * 16 + l16;
            const float* v1 = base + (size_t)(k0 + 16 + 8 * h) * (3 * DMODEL) + 2 * DMODEL + hh * HD + nt * 16 + l16;
            v16bf bf;
            #pragma unroll
            for (int e = 0; e < 8; ++e) bf[e]     = (__bf16)v0[(size_t)e * (3 * DMODEL)];
            #pragma unroll
            for (int e = 0; e < 8; ++e) bf[8 + e] = (__bf16)v1[(size_t)e * (3 * DMODEL)];
            oc[nt] = __builtin_amdgcn_wmma_f32_16x16x32_bf16(
                false, a, false, bf, (short)0, oc[nt], false, false);
        }
    }
    #pragma unroll
    for (int nt = 0; nt < 4; ++nt)
        #pragma unroll
        for (int r = 0; r < 8; ++r) {
            int row = qt * 16 + r + 8 * h;
            out[((size_t)b * S + row) * DMODEL + hh * HD + nt * 16 + l16] = oc[nt][r];
        }
}

// ------------------------------------------------- predict (dot + sigmoid) --
__global__ void predict_kernel(const float* __restrict__ X, const float* __restrict__ w,
                               const float* __restrict__ b, float* __restrict__ sco) {
    int row = blockIdx.x;
    int lane = threadIdx.x;
    const float* xr = X + (size_t)row * DMODEL;
    float s = 0.f;
    for (int d = lane; d < DMODEL; d += 32) s += xr[d] * w[d];
    #pragma unroll
    for (int off = 16; off >= 1; off >>= 1) s += __shfl_xor(s, off, 32);
    if (lane == 0) sco[row] = 1.0f / (1.0f + __expf(-(s + b[0])));
}

// -------------------------------------------------------- stretch kernels ---
__global__ void stretch_score_kernel(const float* __restrict__ sco, const int* __restrict__ seg,
                                     float* __restrict__ out, int T, int S) {
    int i = blockIdx.x * blockDim.x + threadIdx.x;   // over B*T
    int b = i / T;
    out[i] = sco[b * S + seg[i]];
}

__global__ void stretch_add_kernel(const float* __restrict__ orig, const float* __restrict__ feat,
                                   const int* __restrict__ seg, float* __restrict__ out,
                                   int T, int S) {
    size_t i = (size_t)blockIdx.x * blockDim.x + threadIdx.x;  // over B*T*D
    size_t bt = i / DMODEL;
    int d = (int)(i % DMODEL);
    int b = (int)(bt / T);
    int s = seg[bt];
    out[i] = orig[i] + feat[((size_t)b * S + s) * DMODEL + d];
}

// ================================================================= host =====
extern "C" void kernel_launch(void* const* d_in, const int* in_sizes, int n_in,
                              void* d_out, int out_size, void* d_ws, size_t ws_size,
                              hipStream_t stream) {
    (void)out_size; (void)ws_size;
    const int B = 4, T = 1024;

    // ---- locate inputs by unique sizes (robust to pytree flattening order)
    const float* feature = (const float*)d_in[0];
    int sbIdx = -1, shIdx = -1;
    for (int i = 0; i < n_in; ++i) {
        if (in_sizes[i] == B * 31)  sbIdx = i;
        else if (in_sizes[i] == B * 127) shIdx = i;
    }
    if (sbIdx < 0 || shIdx < 0) return;
    const int* scene_bd = (const int*)d_in[sbIdx];
    const int* shot_bd  = (const int*)d_in[shIdx];

    // top-level insertion order puts scene_boundary at index 1; sorted puts it last
    int pstart = (sbIdx == 1) ? 4 : 2;
    // nested sorted-key flattening: first layer tensor is b1 [4096]; insertion: ln1_g [1024]
    bool sorted = (in_sizes[pstart] == 4096);

    struct LayerP { const float *ln1_g,*ln1_b,*w_qkv,*b_qkv,*w_o,*b_o,*ln2_g,*ln2_b,*w1,*b1,*w2,*b2; };
    struct StageP { LayerP layer[2]; const float *pred_w, *pred_b; };
    auto P = [&](int idx) { return (const float*)d_in[pstart + idx]; };
    auto fillStage = [&](StageP& st, int stageIdx) {
        int s0 = stageIdx * 26;
        for (int l = 0; l < 2; ++l) {
            int b0 = s0 + l * 12;
            LayerP& L = st.layer[l];
            if (sorted) {  // b1,b2,b_o,b_qkv,ln1_b,ln1_g,ln2_b,ln2_g,w1,w2,w_o,w_qkv
                L.b1 = P(b0+0);  L.b2 = P(b0+1);  L.b_o = P(b0+2);  L.b_qkv = P(b0+3);
                L.ln1_b = P(b0+4); L.ln1_g = P(b0+5); L.ln2_b = P(b0+6); L.ln2_g = P(b0+7);
                L.w1 = P(b0+8);  L.w2 = P(b0+9);  L.w_o = P(b0+10); L.w_qkv = P(b0+11);
            } else {       // ln1_g,ln1_b,w_qkv,b_qkv,w_o,b_o,ln2_g,ln2_b,w1,b1,w2,b2
                L.ln1_g = P(b0+0); L.ln1_b = P(b0+1); L.w_qkv = P(b0+2); L.b_qkv = P(b0+3);
                L.w_o = P(b0+4);  L.b_o = P(b0+5);  L.ln2_g = P(b0+6); L.ln2_b = P(b0+7);
                L.w1 = P(b0+8);   L.b1 = P(b0+9);   L.w2 = P(b0+10);  L.b2 = P(b0+11);
            }
        }
        if (sorted) { st.pred_b = P(s0 + 24); st.pred_w = P(s0 + 25); }
        else        { st.pred_w = P(s0 + 24); st.pred_b = P(s0 + 25); }
    };
    StageP sc{}, sh{};
    if (sorted) { fillStage(sc, 1); fillStage(sh, 2); }  // frame,scene,shot
    else        { fillStage(sc, 0); fillStage(sh, 1); }  // scene,shot,frame

    // ---- workspace carve-out (~36 MB)
    char* wp = (char*)d_ws;
    auto carve = [&](size_t bytes) { char* p = wp; wp += (bytes + 255) & ~(size_t)255; return p; };
    int*   seg_sc = (int*)  carve((size_t)B * T * sizeof(int));
    int*   seg_sh = (int*)  carve((size_t)B * T * sizeof(int));
    float* scoSeg = (float*)carve((size_t)B * 128 * sizeof(float));
    float* xbuf   = (float*)carve((size_t)B * 128 * DMODEL * sizeof(float));
    float* lnbuf  = (float*)carve((size_t)B * 128 * DMODEL * sizeof(float));
    float* qkvbuf = (float*)carve((size_t)B * 128 * 3 * DMODEL * sizeof(float));
    float* attbuf = (float*)carve((size_t)B * 128 * DMODEL * sizeof(float));
    float* h1buf  = (float*)carve((size_t)B * 128 * 4 * DMODEL * sizeof(float));
    float* feat2  = (float*)carve((size_t)B * T * DMODEL * sizeof(float));

    auto runEncoder = [&](const StageP& st, int S) {
        int M = B * S;
        for (int l = 0; l < 2; ++l) {
            const LayerP& L = st.layer[l];
            layernorm_kernel<<<M, 256, 0, stream>>>(xbuf, L.ln1_g, L.ln1_b, lnbuf);
            gemm_wmma_kernel<<<dim3(3 * DMODEL / 128, M / 64), 256, 0, stream>>>(
                lnbuf, L.w_qkv, L.b_qkv, nullptr, qkvbuf, M, DMODEL, 3 * DMODEL, 0);
            if (S == 32)
                attention_kernel<32><<<dim3(2, NHEAD, B), 32, 0, stream>>>(qkvbuf, attbuf);
            else
                attention_kernel<128><<<dim3(8, NHEAD, B), 32, 0, stream>>>(qkvbuf, attbuf);
            gemm_wmma_kernel<<<dim3(DMODEL / 128, M / 64), 256, 0, stream>>>(
                attbuf, L.w_o, L.b_o, xbuf, xbuf, M, DMODEL, DMODEL, 0);
            layernorm_kernel<<<M, 256, 0, stream>>>(xbuf, L.ln2_g, L.ln2_b, lnbuf);
            gemm_wmma_kernel<<<dim3(4 * DMODEL / 128, M / 64), 256, 0, stream>>>(
                lnbuf, L.w1, L.b1, nullptr, h1buf, M, DMODEL, 4 * DMODEL, 1);
            gemm_wmma_kernel<<<dim3(DMODEL / 128, M / 64), 256, 0, stream>>>(
                h1buf, L.w2, L.b2, xbuf, xbuf, M, 4 * DMODEL, DMODEL, 0);
        }
    };

    float* out = (float*)d_out;

    // ---- scene stage
    seg_kernel<<<B, 256, 0, stream>>>(scene_bd, seg_sc, T, 31);
    seg_kernel<<<B, 256, 0, stream>>>(shot_bd,  seg_sh, T, 127);
    aggregate_kernel<<<dim3(32, B), 256, 0, stream>>>(feature, scene_bd, xbuf, T, 32, 31);
    runEncoder(sc, 32);
    predict_kernel<<<B * 32, 32, 0, stream>>>(xbuf, sc.pred_w, sc.pred_b, scoSeg);
    stretch_score_kernel<<<(B * T) / 256, 256, 0, stream>>>(scoSeg, seg_sc, out, T, 32);

    // ---- shot stage (feature2 = original + stretch(sc_feat))
    stretch_add_kernel<<<(int)(((size_t)B * T * DMODEL) / 256), 256, 0, stream>>>(
        feature, xbuf, seg_sc, feat2, T, 32);
    aggregate_kernel<<<dim3(128, B), 256, 0, stream>>>(feat2, shot_bd, xbuf, T, 128, 127);
    runEncoder(sh, 128);
    predict_kernel<<<B * 128, 32, 0, stream>>>(xbuf, sh.pred_w, sh.pred_b, scoSeg);
    stretch_score_kernel<<<(B * T) / 256, 256, 0, stream>>>(scoSeg, seg_sh, out + B * T, T, 128);

    // frame-level encoder is dead code in the reference (result discarded) -> skipped.
}
